// UnimpNet_44770739093528
// MI455X (gfx1250) — matrix operand order repaired and checked by reading, not compiled, not run on hardware
//
#include <hip/hip_runtime.h>
#include <hip/hip_bf16.h>

typedef __attribute__((ext_vector_type(16))) __bf16 v16bf;
typedef __attribute__((ext_vector_type(8)))  float  v8f;

#define N_NODES 50000
#define N_PAD   50048      // padded to multiple of 128 (8 waves x 16 rows)
#define N_EDGES 800000
#define FEAT    512
#define HEADS   8
#define DHEAD   16
#define INNER   128
#define NCLS    64

// ---------------------------------------------------------------------------
// MaskLabel (x += mask * emb[y]) fused with fp32 -> bf16 conversion
// ---------------------------------------------------------------------------
__global__ void k_masklabel(const float* __restrict__ x,
                            const int* __restrict__ y,
                            const unsigned char* __restrict__ mask,
                            const float* __restrict__ emb,
                            __bf16* __restrict__ xm) {
    long long i = (long long)blockIdx.x * blockDim.x + threadIdx.x;
    if (i >= (long long)N_NODES * FEAT) return;
    int node = (int)(i >> 9);      // /512
    int f    = (int)(i & 511);
    float val = x[i];
    if (mask[node]) val += emb[(long long)y[node] * FEAT + f];
    xm[i] = (__bf16)val;
}

// fp32 [rows, cols] -> bf16 [cols, rows] (transpose so K is contiguous)
__global__ void k_convT(const float* __restrict__ w, __bf16* __restrict__ wt,
                        int rows, int cols) {
    long long i = (long long)blockIdx.x * blockDim.x + threadIdx.x;
    if (i >= (long long)rows * cols) return;
    int r = (int)(i / cols), c = (int)(i % cols);
    wt[(long long)c * rows + r] = (__bf16)w[i];
}

__global__ void k_f2bf(const float* __restrict__ in, __bf16* __restrict__ out,
                       long long n) {
    long long i = (long long)blockIdx.x * blockDim.x + threadIdx.x;
    if (i < n) out[i] = (__bf16)in[i];
}

__global__ void k_init_md(float* __restrict__ m, float* __restrict__ d, int n) {
    int i = blockIdx.x * blockDim.x + threadIdx.x;
    if (i < n) { m[i] = __int_as_float(0xFF800000); d[i] = 0.0f; }
}

// ---------------------------------------------------------------------------
// bf16 WMMA GEMM with async-DMA-staged B panel.
//   Block = 256 threads = 8 waves; block computes a 128-row x (NT*16)-col
//   strip; wave w owns rows [blockIdx*128 + w*16, +16) across all NT col tiles.
//   Per 32-wide k-step the B panel (32 x NT*16 bf16 = NT KB) is staged in LDS
//   via GLOBAL_LOAD_ASYNC_TO_LDS_B128 (memory -> LDS DMA, ASYNCcnt-tracked,
//   no VGPR round trip), then each wave issues NT WMMAs with a register-
//   blocked accumulator strip, reusing its A fragment NT times.
// X : [M_pad, K] row-major bf16.  Wt : [NT*16, K] row-major bf16 (W^T).
// NOTE: Bsh is the only __shared__ object in this kernel, so its static-LDS
// byte offset is 0 and raw LDS addresses can be computed from indices.
// ---------------------------------------------------------------------------
template <int NT>
__global__ void __launch_bounds__(256)
k_gemm_bf16_lds(const __bf16* __restrict__ X, int K,
                const __bf16* __restrict__ Wt,
                const float* __restrict__ bias,
                float* __restrict__ Out, int ldo) {
    __shared__ __align__(16) __bf16 Bsh[NT * 16 * 32];   // [col][k], k fastest

    const int tid  = threadIdx.x;
    const int wave = tid >> 5;
    const int lane = tid & 31;
    const int n    = lane & 15;          // col within tile / row within A tile
    const int half = lane >> 4;          // 0 or 1
    const int R    = blockIdx.x * 128 + wave * 16;

    const __bf16* xrow = X + (long long)(R + n) * K;     // A: row = n

    v8f acc[NT];
#pragma unroll
    for (int t = 0; t < NT; ++t) {
        float bval = bias[t * 16 + n];
#pragma unroll
        for (int r = 0; r < 8; ++r) acc[t][r] = bval;
    }

    const unsigned long long bsh_escape = (unsigned long long)Bsh;

    for (int k0 = 0; k0 < K; k0 += 32) {
        __syncthreads();    // previous panel fully consumed by all waves
        // --- async DMA: stage B panel (NT*16 cols x 32 k) into LDS ---
#pragma unroll
        for (int idx = tid; idx < NT * 64; idx += 256) {
            int col = idx >> 2;                          // 0 .. NT*16-1
            int q   = idx & 3;                           // 16B chunk within col
            unsigned lds_addr = (unsigned)(col * 64 + q * 16);   // bytes (Bsh @ 0)
            unsigned long long gaddr =
                (unsigned long long)(Wt + (long long)col * K + k0 + q * 8);
            asm volatile("global_load_async_to_lds_b128 %0, %1, off"
                         :: "v"(lds_addr), "v"(gaddr), "v"(bsh_escape)
                         : "memory");
        }
        asm volatile("s_wait_asynccnt 0x0" ::: "memory");
        __syncthreads();    // panel visible to all waves

        // --- A fragment (16x32 bf16 layout, ISA 7.12.2) ---
        v16bf av;
#pragma unroll
        for (int j = 0; j < 8; ++j) {
            int ka = k0 + ((j >> 2) << 4) + (half << 3) + ((j & 3) << 1);
            av[2 * j]     = xrow[ka];
            av[2 * j + 1] = xrow[ka + 1];
        }

        // --- NT column tiles: B from LDS, one WMMA each ---
#pragma unroll
        for (int t = 0; t < NT; ++t) {
            const __bf16* bsrc = Bsh + (t * 16 + n) * 32;
            v16bf bvv;
#pragma unroll
            for (int j = 0; j < 8; ++j) {      // B 32x16: vgpr j -> K = (j>>2)*16 + half*8 + 2*(j&3)
                int kb = ((j >> 2) << 4) + (half << 3) + ((j & 3) << 1);
                bvv[2 * j]     = bsrc[kb];
                bvv[2 * j + 1] = bsrc[kb + 1];
            }
            acc[t] = __builtin_amdgcn_wmma_f32_16x16x32_bf16(
                false, av, false, bvv, (short)0, acc[t], false, false);
        }
    }

#pragma unroll
    for (int t = 0; t < NT; ++t) {
#pragma unroll
        for (int r = 0; r < 8; ++r) {
            int row = R + r + half * 8;        // C/D layout: M = vgpr + 8*lanehalf
            if (row < N_NODES)
                Out[(long long)row * ldo + t * 16 + n] = acc[t][r];
        }
    }
}

// ---------------------------------------------------------------------------
// Edge pass 1: logits[e,h] = q[dst]·k[src] / 4 ; atomic max into m[dst,h]
// ---------------------------------------------------------------------------
__global__ void k_edge_logits(const int* __restrict__ src,
                              const int* __restrict__ dst,
                              const float* __restrict__ q,
                              const float* __restrict__ kk,
                              float* __restrict__ logits,
                              float* __restrict__ mbuf) {
    long long i = (long long)blockIdx.x * blockDim.x + threadIdx.x;
    if (i >= (long long)N_EDGES * HEADS) return;
    int e = (int)(i >> 3), h = (int)(i & 7);
    if (e + 2048 < N_EDGES) {                 // stream the index arrays ahead
        __builtin_prefetch(src + e + 2048, 0, 0);
        __builtin_prefetch(dst + e + 2048, 0, 0);
    }
    int s = src[e], d = dst[e];
    const float4* qp = (const float4*)(q  + (long long)d * INNER + h * DHEAD);
    const float4* kp = (const float4*)(kk + (long long)s * INNER + h * DHEAD);
    float dot = 0.f;
#pragma unroll
    for (int j = 0; j < 4; ++j) {
        float4 a = qp[j], b = kp[j];
        dot += a.x * b.x + a.y * b.y + a.z * b.z + a.w * b.w;
    }
    dot *= 0.25f;                        // 1/sqrt(DH=16)
    logits[i] = dot;
    float* addr = mbuf + (long long)d * HEADS + h;
    if (dot >= 0.f) atomicMax((int*)addr, __float_as_int(dot));
    else            atomicMin((unsigned int*)addr, __float_as_uint(dot));
}

// Edge pass 2: p = exp(logit - m[dst]); denom[dst] += p  (p stored in place)
__global__ void k_edge_exp(const int* __restrict__ dst,
                           const float* __restrict__ mbuf,
                           float* __restrict__ logits,
                           float* __restrict__ denom) {
    long long i = (long long)blockIdx.x * blockDim.x + threadIdx.x;
    if (i >= (long long)N_EDGES * HEADS) return;
    int e = (int)(i >> 3), h = (int)(i & 7);
    int d = dst[e];
    float p = __expf(logits[i] - mbuf[(long long)d * HEADS + h]);
    logits[i] = p;
    atomicAdd(denom + (long long)d * HEADS + h, p);
}

// Edge pass 3: agg[dst, c] += (p / denom[dst]) * v[src, c]
__global__ void k_edge_agg(const int* __restrict__ src,
                           const int* __restrict__ dst,
                           const float* __restrict__ p,
                           const float* __restrict__ denom,
                           const float* __restrict__ v,
                           float* __restrict__ agg) {
    long long i = (long long)blockIdx.x * blockDim.x + threadIdx.x;
    if (i >= (long long)N_EDGES * INNER) return;
    long long e = i >> 7;                // /128
    int c = (int)(i & 127);
    int h = c >> 4;
    if (e + 512 < N_EDGES && c == 0) {
        __builtin_prefetch(src + e + 512, 0, 0);
        __builtin_prefetch(dst + e + 512, 0, 0);
    }
    int s = src[e], d = dst[e];
    float alpha = p[e * HEADS + h] / (denom[(long long)d * HEADS + h] + 1e-16f);
    atomicAdd(agg + (long long)d * INNER + c, alpha * v[(long long)s * INNER + c]);
}

// ---------------------------------------------------------------------------
extern "C" void kernel_launch(void* const* d_in, const int* in_sizes, int n_in,
                              void* d_out, int out_size, void* d_ws, size_t ws_size,
                              hipStream_t stream) {
    const float*         x     = (const float*)d_in[0];
    const int*           y     = (const int*)d_in[1];
    const int*           ei    = (const int*)d_in[2];
    const unsigned char* lmask = (const unsigned char*)d_in[3];
    const float*         emb   = (const float*)d_in[4];
    const float* Wq   = (const float*)d_in[5];  const float* bq    = (const float*)d_in[6];
    const float* Wk   = (const float*)d_in[7];  const float* bk    = (const float*)d_in[8];
    const float* Wv   = (const float*)d_in[9];  const float* bv    = (const float*)d_in[10];
    const float* Wsk  = (const float*)d_in[11]; const float* bskip = (const float*)d_in[12];
    const float* Wout = (const float*)d_in[13]; const float* bout  = (const float*)d_in[14];

    const int* src = ei;            // edge_index[0]
    const int* dst = ei + N_EDGES;  // edge_index[1]

    // ---- workspace carve-up (all 256B-aligned; node dim padded to N_PAD) ----
    char* ws = (char*)d_ws;
    size_t off = 0;
    __bf16* xm    = (__bf16*)(ws + off); off += (size_t)N_PAD * FEAT * 2;     // 51.2 MB
    __bf16* wtq   = (__bf16*)(ws + off); off += (size_t)INNER * FEAT * 2;
    __bf16* wtk   = (__bf16*)(ws + off); off += (size_t)INNER * FEAT * 2;
    __bf16* wtv   = (__bf16*)(ws + off); off += (size_t)INNER * FEAT * 2;
    __bf16* wtsk  = (__bf16*)(ws + off); off += (size_t)INNER * FEAT * 2;
    __bf16* wtout = (__bf16*)(ws + off); off += (size_t)NCLS * INNER * 2;
    float* qb   = (float*)(ws + off); off += (size_t)N_PAD * INNER * 4;       // 25.6 MB
    float* kb   = (float*)(ws + off); off += (size_t)N_PAD * INNER * 4;
    float* vb   = (float*)(ws + off); off += (size_t)N_PAD * INNER * 4;
    float* agg  = (float*)(ws + off); off += (size_t)N_PAD * INNER * 4;       // skip + msgs
    float* lg   = (float*)(ws + off); off += (size_t)N_EDGES * HEADS * 4;     // logits -> p
    float* mmax = (float*)(ws + off); off += (size_t)N_NODES * HEADS * 4;
    float* dden = (float*)(ws + off); off += (size_t)N_NODES * HEADS * 4;
    __bf16* hbf = (__bf16*)(ws + off); off += (size_t)N_PAD * INNER * 2;

    // 1) MaskLabel + bf16 conversion of features
    {
        long long n = (long long)N_NODES * FEAT;
        k_masklabel<<<(unsigned)((n + 255) / 256), 256, 0, stream>>>(x, y, lmask, emb, xm);
    }
    // 2) Weight transpose + bf16 conversion
    k_convT<<<(FEAT * INNER + 255) / 256, 256, 0, stream>>>(Wq,   wtq,   FEAT, INNER);
    k_convT<<<(FEAT * INNER + 255) / 256, 256, 0, stream>>>(Wk,   wtk,   FEAT, INNER);
    k_convT<<<(FEAT * INNER + 255) / 256, 256, 0, stream>>>(Wv,   wtv,   FEAT, INNER);
    k_convT<<<(FEAT * INNER + 255) / 256, 256, 0, stream>>>(Wsk,  wtsk,  FEAT, INNER);
    k_convT<<<(INNER * NCLS  + 255) / 256, 256, 0, stream>>>(Wout, wtout, INNER, NCLS);
    // 3) init softmax stats
    k_init_md<<<(N_NODES * HEADS + 255) / 256, 256, 0, stream>>>(mmax, dden, N_NODES * HEADS);

    // 4) q/k/v/skip projections (WMMA bf16, fp32 accumulate, async-LDS B panel)
    const unsigned gblocks = N_PAD / 128;                 // 391
    k_gemm_bf16_lds<8><<<gblocks, 256, 0, stream>>>(xm, FEAT, wtq,  bq,    qb,  INNER);
    k_gemm_bf16_lds<8><<<gblocks, 256, 0, stream>>>(xm, FEAT, wtk,  bk,    kb,  INNER);
    k_gemm_bf16_lds<8><<<gblocks, 256, 0, stream>>>(xm, FEAT, wtv,  bv,    vb,  INNER);
    k_gemm_bf16_lds<8><<<gblocks, 256, 0, stream>>>(xm, FEAT, wtsk, bskip, agg, INNER);

    // 5) edge attention
    {
        long long n1 = (long long)N_EDGES * HEADS;
        k_edge_logits<<<(unsigned)((n1 + 255) / 256), 256, 0, stream>>>(src, dst, qb, kb, lg, mmax);
        k_edge_exp<<<(unsigned)((n1 + 255) / 256), 256, 0, stream>>>(dst, mmax, lg, dden);
        long long n2 = (long long)N_EDGES * INNER;
        k_edge_agg<<<(unsigned)((n2 + 255) / 256), 256, 0, stream>>>(src, dst, lg, dden, vb, agg);
    }

    // 6) output projection: (agg) @ Wout + bout -> d_out [N, 64]
    {
        long long n = (long long)N_NODES * INNER;
        k_f2bf<<<(unsigned)((n + 255) / 256), 256, 0, stream>>>(agg, hbf, n);
        k_gemm_bf16_lds<4><<<gblocks, 256, 0, stream>>>(hbf, INNER, wtout, bout,
                                                        (float*)d_out, NCLS);
    }
}
